// NeuroRNN_52785148068377
// MI455X (gfx1250) — matrix operand order repaired and checked by reading, not compiled, not run on hardware
//
#include <hip/hip_runtime.h>
#include <hip/hip_bf16.h>

// ---------------------------------------------------------------------------
// NeuroRNN on MI455X (gfx1250): persistent-kernel leaky RNN.
//   I_{t+1} = 0.8 I_t + 0.2 (r_t Wrec^T + x_t Win^T)
//   r_{t+1} = 0.9 r_t + 0.1 tanh(I_{t+1})
//   out_t   = r_{t+1} Wout^T
// 64 recurrent WGs (16 H-columns each, weights bf16 in LDS in WMMA B-fragment
// order, I/r state in fp32 VGPRs across all 2048 steps) + 8 output WGs.
// Per-step device-wide split barrier via atomic counters in workspace.
// x is pre-converted to bf16 by a parallel kernel (if ws allows) so the
// serial recurrence does only b128 loads + v_wmma_f32_16x16x32_bf16.
// ---------------------------------------------------------------------------

typedef __bf16 bf16;
typedef bf16  v16bf __attribute__((ext_vector_type(16)));
typedef bf16  v8bf  __attribute__((ext_vector_type(8)));
typedef float v8f   __attribute__((ext_vector_type(8)));

#define B_    32
#define T_    2048
#define IN_   256
#define H_    1024
#define OUT_  128
#define AS_   0.2f
#define AR_   0.1f

#define REC_WGS  (H_ / 16)    // 64
#define OUT_WGS  (OUT_ / 16)  // 8
#define KB_REC   (H_ / 32)    // 32 k-blocks of 32 for r @ Wrec^T
#define KB_IN    (IN_ / 32)   // 8 k-blocks for x @ Win^T
#define ALL_WGS  (REC_WGS + OUT_WGS)

union AFrag { v16bf v; v8bf h[2]; };

__device__ __forceinline__ void wg_wait_ge(unsigned* p, unsigned tgt) {
  if (threadIdx.x == 0) {
    while (__hip_atomic_load(p, __ATOMIC_ACQUIRE, __HIP_MEMORY_SCOPE_AGENT) < tgt)
      __builtin_amdgcn_s_sleep(2);
  }
  __syncthreads();
}

__device__ __forceinline__ void wg_signal(unsigned* p) {
  __syncthreads();  // both waves done with the protected phase
  if (threadIdx.x == 0)
    __hip_atomic_fetch_add(p, 1u, __ATOMIC_RELEASE, __HIP_MEMORY_SCOPE_AGENT);
}

// Parallel one-shot fp32 -> bf16 conversion of x (off the serial path).
__global__ __launch_bounds__(256) void x_to_bf16(const float* __restrict__ x,
                                                 bf16* __restrict__ xbf, int n) {
  const int i = (blockIdx.x * 256 + threadIdx.x) * 8;
  if (i >= n) return;
  const float4 a = *(const float4*)(x + i);
  const float4 b = *(const float4*)(x + i + 4);
  v8bf o;
  o[0] = (bf16)a.x; o[1] = (bf16)a.y; o[2] = (bf16)a.z; o[3] = (bf16)a.w;
  o[4] = (bf16)b.x; o[5] = (bf16)b.y; o[6] = (bf16)b.z; o[7] = (bf16)b.w;
  *(v8bf*)(xbf + i) = o;
}

__global__ __launch_bounds__(64) void neuro_rnn_persistent(
    const float* __restrict__ x,     // [B, T, IN] fp32 (fallback path)
    const bf16*  __restrict__ xbf,   // [B, T, IN] bf16 (fast path; may be null)
    const float* __restrict__ Win,   // [H, IN]
    const float* __restrict__ Wrec,  // [H, H]
    const float* __restrict__ Wout,  // [OUT, H]
    float* __restrict__ out,         // [B, T, OUT]
    unsigned* __restrict__ flags,    // [T]   : #rec WGs that published r_{t+1}
    unsigned* __restrict__ rdone,    // [T+1] : #WGs done reading r_t
    void* __restrict__ rbuf_raw)     // 2 x [B, H] bf16 ping-pong
{
  // LDS weight slices, stored in exact WMMA B-fragment order:
  //   entry (kb, lane) = 16 bf16 = B[k, n] for n = colbase + (lane&15),
  //   k = kb*32 + (lane<16 ? {0..7,16..23} : {8..15,24..31}).  One 32B ds read per lane.
  __shared__ __align__(32) bf16 lds[(KB_REC + KB_IN) * 32 * 16];  // 40 KB

  bf16* rbuf0 = (bf16*)rbuf_raw;
  bf16* rbuf1 = rbuf0 + B_ * H_;

  const int wg   = blockIdx.x;
  const int lane = threadIdx.x & 31;
  const int wave = threadIdx.x >> 5;     // 2 waves: M-tile 0 (rows 0-15), 1 (rows 16-31)
  const int mbase = wave * 16;
  const int lmod  = lane & 15;
  const int khalf = (lane & 16) ? 8 : 0; // A-fragment K sub-offset for upper lanes

  if (wg < REC_WGS) {
    // ------------------------- recurrent workgroup -------------------------
    const int colbase = wg * 16;

    // Wrec slice (B[k][n] = Wrec[n][k]) into fragment-ordered LDS, fp32->bf16.
    for (int e = threadIdx.x; e < KB_REC * 32; e += blockDim.x) {
      const int kb = e >> 5, l = e & 31;
      const int n  = colbase + (l & 15);
      const int k0 = kb * 32 + ((l & 16) ? 8 : 0);
      const float* src = Wrec + (size_t)n * H_ + k0;
      bf16* dst = lds + (size_t)e * 16;
      #pragma unroll
      for (int i = 0; i < 16; ++i) dst[i] = (bf16)src[i + (i & 8)];
    }
    // Win slice (B[k][n] = Win[n][k]).
    for (int e = threadIdx.x; e < KB_IN * 32; e += blockDim.x) {
      const int kb = e >> 5, l = e & 31;
      const int n  = colbase + (l & 15);
      const int k0 = kb * 32 + ((l & 16) ? 8 : 0);
      const float* src = Win + (size_t)n * IN_ + k0;
      bf16* dst = lds + (size_t)(KB_REC * 32 + e) * 16;
      #pragma unroll
      for (int i = 0; i < 16; ++i) dst[i] = (bf16)src[i + (i & 8)];
    }
    __syncthreads();

    // Persistent fp32 state slices [16 rows x 16 cols] in VGPRs.
    v8f Istate, rstate;
    #pragma unroll
    for (int i = 0; i < 8; ++i) { Istate[i] = 0.0f; rstate[i] = 0.0f; }

    const size_t xrow = (size_t)(mbase + lmod) * T_;  // row base in [B,T,IN]

    for (int t = 0; t < T_; ++t) {
      const bf16* rcur  = (t & 1) ? rbuf1 : rbuf0;        // holds r_t
      bf16*       rnext = (t & 1) ? rbuf0 : rbuf1;        // gets r_{t+1}

      if (t > 0) wg_wait_ge(&flags[t - 1], REC_WGS);      // r_t published

      v8f acc;
      #pragma unroll
      for (int i = 0; i < 8; ++i) acc[i] = 0.0f;

      // acc += r_t[16x1024] * WrecT_slice[1024x16]
      const bf16* abase = rcur + (size_t)(mbase + lmod) * H_;
      #pragma unroll 4
      for (int kb = 0; kb < KB_REC; ++kb) {
        AFrag a;
        const int k0 = kb * 32 + khalf;
        a.h[0] = *(const v8bf*)(abase + k0);
        a.h[1] = *(const v8bf*)(abase + k0 + 16);
        const v16bf b = *(const v16bf*)(lds + (size_t)(kb * 32 + lane) * 16);
        acc = __builtin_amdgcn_wmma_f32_16x16x32_bf16(
                  false, a.v, false, b, (short)0, acc, false, false);
      }
      // acc += x_t[16x256] * WinT_slice[256x16]
      if (xbf) {
        // Fast path: x already bf16, pure b128 loads.
        const bf16* xb = xbf + (xrow + t) * IN_;
        #pragma unroll 4
        for (int kb = 0; kb < KB_IN; ++kb) {
          AFrag a;
          const int k0 = kb * 32 + khalf;
          a.h[0] = *(const v8bf*)(xb + k0);
          a.h[1] = *(const v8bf*)(xb + k0 + 16);
          const v16bf b = *(const v16bf*)(lds + (size_t)((KB_REC + kb) * 32 + lane) * 16);
          acc = __builtin_amdgcn_wmma_f32_16x16x32_bf16(
                    false, a.v, false, b, (short)0, acc, false, false);
        }
      } else {
        // Fallback: convert fp32 -> bf16 on the fly.
        const float* xb = x + (xrow + t) * IN_;
        #pragma unroll 2
        for (int kb = 0; kb < KB_IN; ++kb) {
          AFrag a;
          const int k0 = kb * 32 + khalf;
          #pragma unroll
          for (int i = 0; i < 8; ++i) a.v[i]     = (bf16)xb[k0 + i];
          #pragma unroll
          for (int i = 0; i < 8; ++i) a.v[8 + i] = (bf16)xb[k0 + 16 + i];
          const v16bf b = *(const v16bf*)(lds + (size_t)((KB_REC + kb) * 32 + lane) * 16);
          acc = __builtin_amdgcn_wmma_f32_16x16x32_bf16(
                    false, a.v, false, b, (short)0, acc, false, false);
        }
      }

      wg_signal(&rdone[t]);   // done reading r_t

      // Leaky state update, fp32, tanh on the fly.
      #pragma unroll
      for (int i = 0; i < 8; ++i) {
        Istate[i] = (1.0f - AS_) * Istate[i] + AS_ * acc[i];
        rstate[i] = (1.0f - AR_) * rstate[i] + AR_ * tanhf(Istate[i]);
      }

      // WAR guard: rnext currently holds r_{t-1}; all its readers must finish.
      if (t >= 1)
        wg_wait_ge(&rdone[t - 1], (t == 1) ? REC_WGS : ALL_WGS);

      // Publish r_{t+1} slice (D-fragment layout: VGPR i -> row i / i+8).
      {
        const int n  = colbase + lmod;
        const int m0 = mbase + ((lane & 16) ? 8 : 0);
        #pragma unroll
        for (int i = 0; i < 8; ++i)
          rnext[(size_t)(m0 + i) * H_ + n] = (bf16)rstate[i];
      }
      __threadfence();
      wg_signal(&flags[t]);
    }
  } else {
    // -------------------------- output workgroup ---------------------------
    const int colbase = (wg - REC_WGS) * 16;   // OUT columns

    // Wout slice (B[k][n] = Wout[n][k]) into fragment-ordered LDS.
    for (int e = threadIdx.x; e < KB_REC * 32; e += blockDim.x) {
      const int kb = e >> 5, l = e & 31;
      const int n  = colbase + (l & 15);
      const int k0 = kb * 32 + ((l & 16) ? 8 : 0);
      const float* src = Wout + (size_t)n * H_ + k0;
      bf16* dst = lds + (size_t)e * 16;
      #pragma unroll
      for (int i = 0; i < 16; ++i) dst[i] = (bf16)src[i + (i & 8)];
    }
    __syncthreads();

    for (int t = 0; t < T_; ++t) {
      const bf16* rv = ((t + 1) & 1) ? rbuf1 : rbuf0;     // holds r_{t+1}
      wg_wait_ge(&flags[t], REC_WGS);

      v8f acc;
      #pragma unroll
      for (int i = 0; i < 8; ++i) acc[i] = 0.0f;

      const bf16* abase = rv + (size_t)(mbase + lmod) * H_;
      #pragma unroll 4
      for (int kb = 0; kb < KB_REC; ++kb) {
        AFrag a;
        const int k0 = kb * 32 + khalf;
        a.h[0] = *(const v8bf*)(abase + k0);
        a.h[1] = *(const v8bf*)(abase + k0 + 16);
        const v16bf b = *(const v16bf*)(lds + (size_t)(kb * 32 + lane) * 16);
        acc = __builtin_amdgcn_wmma_f32_16x16x32_bf16(
                  false, a.v, false, b, (short)0, acc, false, false);
      }

      wg_signal(&rdone[t + 1]);  // done reading r_{t+1}

      // out[b, t, o], fp32.
      const int o  = colbase + lmod;
      const int m0 = mbase + ((lane & 16) ? 8 : 0);
      #pragma unroll
      for (int i = 0; i < 8; ++i)
        out[((size_t)(m0 + i) * T_ + t) * OUT_ + o] = acc[i];
    }
  }
}

extern "C" void kernel_launch(void* const* d_in, const int* in_sizes, int n_in,
                              void* d_out, int out_size, void* d_ws, size_t ws_size,
                              hipStream_t stream) {
  (void)in_sizes; (void)n_in; (void)out_size;
  const float* x    = (const float*)d_in[0];  // [B, T, IN]
  const float* Win  = (const float*)d_in[1];  // [H, IN]
  const float* Wrec = (const float*)d_in[2];  // [H, H]
  const float* Wout = (const float*)d_in[3];  // [OUT, H]
  float* out = (float*)d_out;                 // [B, T, OUT]

  // Workspace layout:
  //   [flags 16KB][rdone 16KB][rbuf0 64KB][rbuf1 64KB][xbf 32MB (optional)]
  const size_t fixed_bytes = (160u << 10);
  const size_t xelems = (size_t)B_ * T_ * IN_;          // 16,777,216
  const size_t xbytes = xelems * sizeof(bf16);          // 32 MB

  unsigned* flags = (unsigned*)d_ws;
  unsigned* rdone = (unsigned*)((char*)d_ws + (16u << 10));
  void*     rbuf  = (void*)((char*)d_ws + (32u << 10));

  bf16* xbf = nullptr;
  if (ws_size >= fixed_bytes + xbytes)
    xbf = (bf16*)((char*)d_ws + fixed_bytes);

  // Zero flags, rdone, and rbuf0 (r_0 = 0). Graph-capturable; replays re-init.
  hipMemsetAsync(d_ws, 0, (32u << 10) + (size_t)B_ * H_ * sizeof(bf16), stream);

  if (xbf) {
    const int n = (int)xelems;
    x_to_bf16<<<(n / 8 + 255) / 256, 256, 0, stream>>>(x, xbf, n);
  }

  neuro_rnn_persistent<<<ALL_WGS, 64, 0, stream>>>(
      x, xbf, Win, Wrec, Wout, out, flags, rdone, rbuf);
}
